// StructureSetAbstractionMsg_12506944766651
// MI455X (gfx1250) — matrix operand order repaired and checked by reading, not compiled, not run on hardware
//
#include <hip/hip_runtime.h>
#include <hip/hip_bf16.h>
#include <cstdint>

// PointNet++ MSG set abstraction for MI455X (gfx1250, wave32, WMMA).
// Channel-last f16 activations; double-buffered LDS GEMM tiles; full-K W tile
// staged once; 4 WMMA per wave per K-chunk; vectorized streaming epilogues.

typedef __attribute__((ext_vector_type(16))) _Float16 v16h;
typedef __attribute__((ext_vector_type(8)))  _Float16 v8h;
typedef __attribute__((ext_vector_type(8)))  float    v8f;

#define BDIM 8
#define NPTS 4096
#define SPTS 512
#define INCH 64
#define CTOT 640
#define MT   256  // M tile per block
#define XSTR 40   // LDS column stride in halves (80B, 16B-aligned)

#define WMMA_F16(a, b, c) \
  __builtin_amdgcn_wmma_f32_16x16x32_f16(false, (a), false, (b), (short)0, (c), false, false)

// ---- WMMA fragment loaders (ISA 7.12.2 layouts) -----------------------------

// A: 16x32 slice of an LDS row-major [16+][rstride] tile at column k0.
// lanes 0-15 : row=lane,    halves = K{0..7}   then K{16..23}
// lanes 16-31: row=lane-16, halves = K{8..15}  then K{24..31}
__device__ inline v16h load_a_frag(const _Float16* Wt, int rstride, int k0, int lane) {
  int row  = lane & 15;
  int koff = k0 + ((lane >> 4) << 3);
  union { v16h v; v8h h2[2]; } u;
  const _Float16* p = Wt + row * rstride + koff;
  u.h2[0] = *(const v8h*)(p);
  u.h2[1] = *(const v8h*)(p + 16);
  return u.v;
}

// B: 32x16 f16 tile staged column-major in LDS: Xt[col][k], stride XSTR halves.
// lanes 0-15 : col=lane,    K{0..7} then K{16..23} ; lanes 16-31: col=lane-16, K{8..15} then K{24..31}
__device__ inline v16h load_b_frag(const _Float16* Xt, int colbase, int lane) {
  int col  = colbase + (lane & 15);
  int koff = (lane >> 4) << 3;
  union { v16h v; v8h h2[2]; } u;
  const _Float16* p = Xt + col * XSTR + koff;
  u.h2[0] = *(const v8h*)(p);
  u.h2[1] = *(const v8h*)(p + 16);
  return u.v;
}

// ---- 1) Ball query: one wave per (b,s), all three radii in one scan ---------

__global__ void ball_query_kernel(const float* __restrict__ xyz,
                                  const float* __restrict__ sxyz,
                                  int* __restrict__ o1, int* __restrict__ o2,
                                  int* __restrict__ o3) {
  int gtid = blockIdx.x * blockDim.x + threadIdx.x;
  int wave = gtid >> 5;
  int lane = gtid & 31;
  if (wave >= BDIM * SPTS) return;
  int b = wave / SPTS, s = wave % SPTS;
  const float* c = sxyz + (size_t)(b * SPTS + s) * 3;
  float cx = c[0], cy = c[1], cz = c[2];
  int* q1 = o1 + (size_t)(b * SPTS + s) * 16;
  int* q2 = o2 + (size_t)(b * SPTS + s) * 32;
  int* q3 = o3 + (size_t)(b * SPTS + s) * 64;
  int c1 = 0, c2 = 0, c3 = 0, f1 = 0, f2 = 0, f3 = 0;
  const float r1 = 0.01f, r2 = 0.04f, r3 = 0.16f;  // squared radii
  unsigned long long lt = (1ull << lane) - 1ull;
  for (int base = 0; base < NPTS; base += 32) {
    int n = base + lane;
    const float* p = xyz + ((size_t)b * NPTS + n) * 3;
    float dx = p[0] - cx, dy = p[1] - cy, dz = p[2] - cz;
    float d2 = dx * dx + dy * dy + dz * dz;
    unsigned long long m1 = __ballot(d2 <= r1);
    unsigned long long m2 = __ballot(d2 <= r2);
    unsigned long long m3 = __ballot(d2 <= r3);
    if (m1 && c1 == 0) f1 = base + (int)__ffsll(m1) - 1;
    if (m2 && c2 == 0) f2 = base + (int)__ffsll(m2) - 1;
    if (m3 && c3 == 0) f3 = base + (int)__ffsll(m3) - 1;
    if (((m1 >> lane) & 1) && c1 < 16) { int p1 = c1 + __popcll(m1 & lt); if (p1 < 16) q1[p1] = n; }
    if (((m2 >> lane) & 1) && c2 < 32) { int p2 = c2 + __popcll(m2 & lt); if (p2 < 32) q2[p2] = n; }
    if (((m3 >> lane) & 1) && c3 < 64) { int p3 = c3 + __popcll(m3 & lt); if (p3 < 64) q3[p3] = n; }
    c1 += __popcll(m1); c2 += __popcll(m2); c3 += __popcll(m3);
    if (c1 >= 16 && c2 >= 32 && c3 >= 64) break;   // uniform across wave
  }
  if (c1 > 16) c1 = 16;  if (c2 > 32) c2 = 32;  if (c3 > 64) c3 = 64;
  for (int p = c1 + lane; p < 16; p += 32) q1[p] = f1;
  for (int p = c2 + lane; p < 32; p += 32) q2[p] = f2;
  for (int p = c3 + lane; p < 64; p += 32) q3[p] = f3;
}

// ---- 2) Gather: X[m][0..95] = {points[b,idx,:64], dxyz[3], zeros} (f16) -----

__global__ void gather_kernel(const float* __restrict__ xyz,
                              const float* __restrict__ sxyz,
                              const float* __restrict__ points,
                              const int* __restrict__ idx,
                              _Float16* __restrict__ X, int k) {
  int m = blockIdx.x * blockDim.x + threadIdx.x;
  int Mtot = BDIM * SPTS * k;
  if (m >= Mtot) return;
  int bs = m / k;
  int b = bs / SPTS, s = bs % SPTS;
  int id = idx[m];
  const float* src = points + ((size_t)b * NPTS + id) * INCH;
  _Float16* dst = X + (size_t)m * 96;
  union { _Float16 h[2]; unsigned u; } pk;
#pragma unroll
  for (int c = 0; c < INCH; c += 2) {
    pk.h[0] = (_Float16)src[c];
    pk.h[1] = (_Float16)src[c + 1];
    *(unsigned*)(dst + c) = pk.u;
  }
  const float* pxyz = xyz + ((size_t)b * NPTS + id) * 3;
  const float* cwq  = sxyz + ((size_t)(b * SPTS + s)) * 3;
  dst[64] = (_Float16)(pxyz[0] - cwq[0]);
  dst[65] = (_Float16)(pxyz[1] - cwq[1]);
  dst[66] = (_Float16)(pxyz[2] - cwq[2]);
  dst[67] = (_Float16)0.f;
  pk.h[0] = (_Float16)0.f; pk.h[1] = (_Float16)0.f;
#pragma unroll
  for (int c = 68; c < 96; c += 2) *(unsigned*)(dst + c) = pk.u;
}

// ---- 2b) Pack weights: f32 [C_out][C_in] -> f16 [C_out][Kpad], zero-padded --

__global__ void pack_w_kernel(const float* __restrict__ W, _Float16* __restrict__ Wp,
                              int C_out, int C_in, int Kpad) {
  int t = blockIdx.x * blockDim.x + threadIdx.x;
  if (t >= C_out * Kpad) return;
  int c = t % Kpad, r = t / Kpad;
  Wp[t] = (c < C_in) ? (_Float16)W[(size_t)r * C_in + c] : (_Float16)0.f;
}

// ---- 3) GEMM (WMMA f16->f32) + bias + per-channel sum/sumsq atomics ---------
// Block = 256 thr = 8 waves; tile 32(C_out) x 256(M); 4 WMMA / wave / chunk.
// Full-K W tile staged once; double-buffered X tile (overlap loads w/ WMMA).

__global__ __launch_bounds__(256)
void gemm_bn_stats_kernel(const _Float16* __restrict__ Wp, const float* __restrict__ bias,
                          const _Float16* __restrict__ X, _Float16* __restrict__ Y,
                          float* __restrict__ stats, int C_out, int K, int M) {
  __shared__ _Float16 Wt[32 * 128];        // full-K tile for 32 output channels
  __shared__ _Float16 Xt[2][MT * XSTR];    // double-buffered 256x32 X tile
  int tid = threadIdx.x;
  int wavei = tid >> 5, lane = tid & 31;
  int m0    = blockIdx.x * MT;
  int cout0 = blockIdx.y * 32;
  // stage W (32 x K) once (K multiple of 32, rows stay v8h-aligned)
  for (int idx = tid * 8; idx < 32 * K; idx += 256 * 8) {
    int r = idx / K, c = idx % K;
    *(v8h*)(Wt + idx) = *(const v8h*)(Wp + (size_t)(cout0 + r) * K + c);
  }
  const _Float16* xrow = X + (size_t)(m0 + tid) * K;
  {
    _Float16* dst = Xt[0] + tid * XSTR;
    *(v8h*)(dst)      = *(const v8h*)(xrow);
    *(v8h*)(dst + 8)  = *(const v8h*)(xrow + 8);
    *(v8h*)(dst + 16) = *(const v8h*)(xrow + 16);
    *(v8h*)(dst + 24) = *(const v8h*)(xrow + 24);
  }
  __syncthreads();
  v8f acc00 = {}, acc01 = {}, acc10 = {}, acc11 = {};
  int nc = K / 32;
  for (int c = 0; c < nc; ++c) {
    v8h p0, p1, p2, p3;
    bool pref = (c + 1 < nc);
    if (pref) {  // issue next chunk's global loads early (overlap with WMMA)
      const _Float16* src = xrow + (c + 1) * 32;
      p0 = *(const v8h*)(src);
      p1 = *(const v8h*)(src + 8);
      p2 = *(const v8h*)(src + 16);
      p3 = *(const v8h*)(src + 24);
    }
    const _Float16* cur = Xt[c & 1];
    v16h a0 = load_a_frag(Wt, K, c * 32, lane);
    v16h a1 = load_a_frag(Wt + 16 * K, K, c * 32, lane);
    v16h b0 = load_b_frag(cur, wavei * 32, lane);
    v16h b1 = load_b_frag(cur, wavei * 32 + 16, lane);
    acc00 = WMMA_F16(a0, b0, acc00);
    acc01 = WMMA_F16(a0, b1, acc01);
    acc10 = WMMA_F16(a1, b0, acc10);
    acc11 = WMMA_F16(a1, b1, acc11);
    if (pref) {
      _Float16* dst = Xt[(c + 1) & 1] + tid * XSTR;
      *(v8h*)(dst)      = p0;
      *(v8h*)(dst + 8)  = p1;
      *(v8h*)(dst + 16) = p2;
      *(v8h*)(dst + 24) = p3;
    }
    __syncthreads();
  }
  // Epilogue: bias, 16B f16 stores, BN stat atomics.
  int halfsel = lane >> 4;
  int chb0 = cout0 + 8 * halfsel;
  int chb1 = chb0 + 16;
  int mA = m0 + wavei * 32 + (lane & 15);
  int mB = mA + 16;
  float bv0[8], bv1[8];
#pragma unroll
  for (int r = 0; r < 8; ++r) { bv0[r] = bias[chb0 + r]; bv1[r] = bias[chb1 + r]; }
  float y00[8], y01[8], y10[8], y11[8];
  union { _Float16 h[8]; v8h v; } o;
#pragma unroll
  for (int r = 0; r < 8; ++r) { y00[r] = acc00[r] + bv0[r]; o.h[r] = (_Float16)y00[r]; }
  *(v8h*)(Y + (size_t)mA * C_out + chb0) = o.v;
#pragma unroll
  for (int r = 0; r < 8; ++r) { y01[r] = acc01[r] + bv0[r]; o.h[r] = (_Float16)y01[r]; }
  *(v8h*)(Y + (size_t)mB * C_out + chb0) = o.v;
#pragma unroll
  for (int r = 0; r < 8; ++r) { y10[r] = acc10[r] + bv1[r]; o.h[r] = (_Float16)y10[r]; }
  *(v8h*)(Y + (size_t)mA * C_out + chb1) = o.v;
#pragma unroll
  for (int r = 0; r < 8; ++r) { y11[r] = acc11[r] + bv1[r]; o.h[r] = (_Float16)y11[r]; }
  *(v8h*)(Y + (size_t)mB * C_out + chb1) = o.v;
#pragma unroll
  for (int r = 0; r < 8; ++r) {
    float s0 = y00[r] + y01[r], q0 = y00[r] * y00[r] + y01[r] * y01[r];
    float s1 = y10[r] + y11[r], q1 = y10[r] * y10[r] + y11[r] * y11[r];
#pragma unroll
    for (int mm = 1; mm < 16; mm <<= 1) {
      s0 += __shfl_xor(s0, mm); q0 += __shfl_xor(q0, mm);
      s1 += __shfl_xor(s1, mm); q1 += __shfl_xor(q1, mm);
    }
    if ((lane & 15) == 0) {
      atomicAdd(&stats[chb0 + r], s0);
      atomicAdd(&stats[C_out + chb0 + r], q0);
      atomicAdd(&stats[chb1 + r], s1);
      atomicAdd(&stats[C_out + chb1 + r], q1);
    }
  }
}

// ---- 4) BatchNorm (batch stats) + ReLU, vectorized in-place f16 -------------

__global__ void bn_relu_kernel(_Float16* __restrict__ Y, const float* __restrict__ stats,
                               const float* __restrict__ g, const float* __restrict__ bt,
                               int C_out, int M) {
  size_t total8 = (size_t)C_out * M / 8;
  size_t t = (size_t)blockIdx.x * blockDim.x + threadIdx.x;
  if (t >= total8) return;
  size_t base = t * 8;
  int c0 = (int)(base % (size_t)C_out);   // 8 | C_out: chunk stays in one row
  float invM = 1.f / (float)M;
  union { _Float16 h[8]; v8h v; } u;
  u.v = *(const v8h*)(Y + base);
#pragma unroll
  for (int j = 0; j < 8; ++j) {
    int c = c0 + j;
    float mu  = stats[c] * invM;
    float var = stats[C_out + c] * invM - mu * mu;
    float y = (float)u.h[j];
    y = g[c] * (y - mu) * rsqrtf(var + 1e-5f) + bt[c];
    u.h[j] = (_Float16)fmaxf(y, 0.f);
  }
  *(v8h*)(Y + base) = u.v;
}

// ---- 5) Max over k samples -> feats [B][S][640] (f32), vectorized -----------

__global__ void maxpool_kernel(const _Float16* __restrict__ Y, float* __restrict__ feats,
                               int C_out, int k, int c_off) {
  int t = blockIdx.x * blockDim.x + threadIdx.x;
  int cw = C_out / 8;
  int total = BDIM * SPTS * cw;
  if (t >= total) return;
  int c0 = (t % cw) * 8;
  int bs = t / cw;
  const _Float16* p = Y + (size_t)bs * k * C_out + c0;
  float mx[8];
#pragma unroll
  for (int r = 0; r < 8; ++r) mx[r] = -3.0e38f;
  for (int j = 0; j < k; ++j) {
    union { _Float16 h[8]; v8h v; } u;
    u.v = *(const v8h*)(p + (size_t)j * C_out);
#pragma unroll
    for (int r = 0; r < 8; ++r) mx[r] = fmaxf(mx[r], (float)u.h[r]);
  }
  float* o = feats + (size_t)bs * CTOT + c_off + c0;
#pragma unroll
  for (int r = 0; r < 8; ++r) o[r] = mx[r];
}

// ---- 6) Row L2-normalize -> fnT [B][S][640] (f16), wave per row -------------

__global__ void normalize_kernel(const float* __restrict__ feats,
                                 _Float16* __restrict__ fnT) {
  int gtid = blockIdx.x * blockDim.x + threadIdx.x;
  int wave = gtid >> 5, lane = gtid & 31;
  if (wave >= BDIM * SPTS) return;
  const float* row = feats + (size_t)wave * CTOT;
  float sum = 0.f;
  for (int i = lane; i < CTOT; i += 32) { float v = row[i]; sum += v * v; }
#pragma unroll
  for (int mm = 16; mm >= 1; mm >>= 1) sum += __shfl_xor(sum, mm);
  float inv = 1.f / fmaxf(sqrtf(sum), 1e-8f);
  _Float16* o = fnT + (size_t)wave * CTOT;
  for (int i = lane; i < CTOT; i += 32) o[i] = (_Float16)(row[i] * inv);
}

// ---- 7) Gram: out[b] = fnT . fnT^T (512x512x640), WMMA, double-buffered -----

__global__ __launch_bounds__(256)
void gram_kernel(const _Float16* __restrict__ fnT, float* __restrict__ out) {
  __shared__ _Float16 At[16 * CTOT];       // full-K A tile (20KB)
  __shared__ _Float16 Bt[2][MT * XSTR];    // double-buffered B tile (40KB)
  int tid = threadIdx.x;
  int wavei = tid >> 5, lane = tid & 31;
  int j0 = blockIdx.x * MT;
  int i0 = blockIdx.y * 16;
  int b  = blockIdx.z;
  for (int idx = tid * 8; idx < 16 * CTOT; idx += 256 * 8) {
    int r = idx / CTOT, c = idx % CTOT;    // 8 | 640: chunks stay in-row
    *(v8h*)(At + idx) = *(const v8h*)(fnT + ((size_t)(b * SPTS + i0 + r)) * CTOT + c);
  }
  const _Float16* brow = fnT + ((size_t)(b * SPTS + j0 + tid)) * CTOT;
  {
    _Float16* dst = Bt[0] + tid * XSTR;
    *(v8h*)(dst)      = *(const v8h*)(brow);
    *(v8h*)(dst + 8)  = *(const v8h*)(brow + 8);
    *(v8h*)(dst + 16) = *(const v8h*)(brow + 16);
    *(v8h*)(dst + 24) = *(const v8h*)(brow + 24);
  }
  __syncthreads();
  v8f acc0 = {}, acc1 = {};
  const int nc = CTOT / 32;
  for (int c = 0; c < nc; ++c) {
    v8h p0, p1, p2, p3;
    bool pref = (c + 1 < nc);
    if (pref) {
      const _Float16* src = brow + (c + 1) * 32;
      p0 = *(const v8h*)(src);
      p1 = *(const v8h*)(src + 8);
      p2 = *(const v8h*)(src + 16);
      p3 = *(const v8h*)(src + 24);
    }
    const _Float16* cur = Bt[c & 1];
    v16h a  = load_a_frag(At, CTOT, c * 32, lane);
    v16h b0 = load_b_frag(cur, wavei * 32, lane);
    v16h b1 = load_b_frag(cur, wavei * 32 + 16, lane);
    acc0 = WMMA_F16(a, b0, acc0);
    acc1 = WMMA_F16(a, b1, acc1);
    if (pref) {
      _Float16* dst = Bt[(c + 1) & 1] + tid * XSTR;
      *(v8h*)(dst)      = p0;
      *(v8h*)(dst + 8)  = p1;
      *(v8h*)(dst + 16) = p2;
      *(v8h*)(dst + 24) = p3;
    }
    __syncthreads();
  }
  size_t base = (size_t)b * SPTS * SPTS;
  int col0 = j0 + wavei * 32 + (lane & 15);
#pragma unroll
  for (int r = 0; r < 8; ++r) {
    int row = i0 + r + 8 * (lane >> 4);
    out[base + (size_t)row * SPTS + col0]      = acc0[r];
    out[base + (size_t)row * SPTS + col0 + 16] = acc1[r];
  }
}

// ---- Host orchestration -----------------------------------------------------

extern "C" void kernel_launch(void* const* d_in, const int* in_sizes, int n_in,
                              void* d_out, int out_size, void* d_ws, size_t ws_size,
                              hipStream_t stream) {
  (void)in_sizes; (void)n_in; (void)out_size; (void)ws_size;
  const float* xyz  = (const float*)d_in[0];
  const float* sxyz = (const float*)d_in[1];
  const float* pts  = (const float*)d_in[2];
  const float* conv_w[3][3]; const float* conv_b[3][3];
  const float* bn_g[3][3];   const float* bn_b[3][3];
  for (int i = 0; i < 3; ++i)
    for (int l = 0; l < 3; ++l) {
      conv_w[i][l] = (const float*)d_in[3  + i * 3 + l];
      conv_b[i][l] = (const float*)d_in[12 + i * 3 + l];
      bn_g[i][l]   = (const float*)d_in[21 + i * 3 + l];
      bn_b[i][l]   = (const float*)d_in[30 + i * 3 + l];
    }

  uint8_t* ws = (uint8_t*)d_ws;
  size_t off = 0;
  auto alloc = [&](size_t bytes) -> void* {
    void* p = ws + off;
    off += (bytes + 255) & ~(size_t)255;
    return p;
  };
  int*   idx1  = (int*)alloc((size_t)BDIM * SPTS * 16 * sizeof(int));
  int*   idx2  = (int*)alloc((size_t)BDIM * SPTS * 32 * sizeof(int));
  int*   idx3  = (int*)alloc((size_t)BDIM * SPTS * 64 * sizeof(int));
  float* stats = (float*)alloc(512 * sizeof(float));
  float* feats = (float*)alloc((size_t)BDIM * SPTS * CTOT * sizeof(float));
  _Float16* fnT = (_Float16*)alloc((size_t)BDIM * SPTS * CTOT * sizeof(_Float16));
  _Float16* Wp  = (_Float16*)alloc((size_t)256 * 128 * sizeof(_Float16));
  size_t Mmax = (size_t)BDIM * SPTS * 64;
  _Float16* bufA = (_Float16*)alloc(Mmax * 128 * sizeof(_Float16));  // X side (K<=128)
  _Float16* bufB = (_Float16*)alloc(Mmax * 256 * sizeof(_Float16));  // Y side (C<=256)

  ball_query_kernel<<<(BDIM * SPTS * 32 + 255) / 256, 256, 0, stream>>>(
      xyz, sxyz, idx1, idx2, idx3);

  const int  ks[3]       = {16, 32, 64};
  const int  couts[3][3] = {{64, 64, 128}, {128, 128, 256}, {128, 128, 256}};
  const int  cins[3][3]  = {{67, 64, 64},  {67, 128, 128},  {67, 128, 128}};
  const int* idxs[3]     = {idx1, idx2, idx3};
  int c_off = 0;
  for (int i = 0; i < 3; ++i) {
    int k = ks[i];
    int M = BDIM * SPTS * k;
    gather_kernel<<<(M + 255) / 256, 256, 0, stream>>>(xyz, sxyz, pts, idxs[i], bufA, k);

    _Float16* Xc = bufA;
    _Float16* Yc = bufB;
    int Kcur = 96;  // padded input channel count (67 -> 96)
    for (int l = 0; l < 3; ++l) {
      int C_out = couts[i][l];
      pack_w_kernel<<<(C_out * Kcur + 255) / 256, 256, 0, stream>>>(
          conv_w[i][l], Wp, C_out, cins[i][l], Kcur);
      hipMemsetAsync(stats, 0, 2 * C_out * sizeof(float), stream);
      dim3 grid(M / MT, C_out / 32);
      gemm_bn_stats_kernel<<<grid, 256, 0, stream>>>(
          Wp, conv_b[i][l], Xc, Yc, stats, C_out, Kcur, M);
      size_t tot8 = (size_t)C_out * M / 8;
      bn_relu_kernel<<<(int)((tot8 + 255) / 256), 256, 0, stream>>>(
          Yc, stats, bn_g[i][l], bn_b[i][l], C_out, M);
      _Float16* t = Xc; Xc = Yc; Yc = t;   // ping-pong
      Kcur = C_out;
    }
    int C_fin = couts[i][2];
    int tot2 = BDIM * SPTS * (C_fin / 8);
    maxpool_kernel<<<(tot2 + 255) / 256, 256, 0, stream>>>(Xc, feats, C_fin, k, c_off);
    c_off += C_fin;
  }

  normalize_kernel<<<(BDIM * SPTS * 32 + 255) / 256, 256, 0, stream>>>(feats, fnT);

  dim3 ggrid(SPTS / MT, SPTS / 16, BDIM);
  gram_kernel<<<ggrid, 256, 0, stream>>>(fnT, (float*)d_out);
}
